// ScaleDotProductAttention_17738214933179
// MI455X (gfx1250) — matrix operand order repaired and checked by reading, not compiled, not run on hardware
//
#include <hip/hip_runtime.h>

typedef __attribute__((ext_vector_type(16))) __bf16 v16bf;
typedef __attribute__((ext_vector_type(8)))  float  v8f;
typedef __attribute__((ext_vector_type(4)))  float  f4;

#define B_      4
#define H_      16
#define L_      2048
#define D_      128
#define QTILE   16
#define NWAVES  8
#define THREADS 256
#define SSTRIDE (L_ + 4)   // padded row stride (floats): rotates LDS banks per row,
                           // keeps 16B alignment for b128 LDS/global accesses

__global__ __launch_bounds__(THREADS)
void sdpa_bf16_wmma_kernel(const float* __restrict__ qg,
                           const float* __restrict__ kg,
                           const float* __restrict__ vg,
                           const int*   __restrict__ maskg,
                           float* __restrict__ outg,    // [B,H,L,D]
                           float* __restrict__ probsg)  // [B,H,L,L]
{
    extern __shared__ float lds[];
    float* s_scores = lds;                          // 16 * SSTRIDE f32 (~128 KB)
    float* s_red    = lds + QTILE * SSTRIDE;        // 16 * 16 partials
    float* s_stat   = s_red + QTILE * 16;           // [0..15]=rowmax, [16..31]=1/rowsum

    const int tid  = threadIdx.x;
    const int lane = tid & 31;
    const int wave = tid >> 5;
    const int half = lane >> 4;                     // half-wave select
    const int l16  = lane & 15;

    const int gid = blockIdx.x;
    const int qb  = gid & (L_ / QTILE - 1);
    const int hb  = gid / (L_ / QTILE);
    const int h   = hb & (H_ - 1);
    const int b   = hb / H_;
    const int q0  = qb * QTILE;

    const float scale = 0.08838834764831845f;       // 1/sqrt(128)

    const size_t head = (size_t)(b * H_ + h);
    const float* Q = qg + head * L_ * D_ + (size_t)q0 * D_;
    const float* K = kg + head * L_ * D_;
    const float* V = vg + head * L_ * D_;
    const int*   M = maskg + (size_t)b * L_ * L_ + (size_t)q0 * L_;

    // ---------------- Q A-fragments (reused across whole k loop) ------------
    // A 16x32 bf16: lanes 0-15 -> M=lane, K {0..15}; lanes 16-31 -> K {16..31}
    v16bf a_q[4];
    {
        const float* qrow = Q + (size_t)l16 * D_;
        #pragma unroll
        for (int c = 0; c < 4; ++c) {
            const int d0 = c * 32 + 16 * half;
            #pragma unroll
            for (int i = 0; i < 8; ++i) {
                a_q[c][i]     = (__bf16)qrow[d0 + i];
                a_q[c][i + 8] = (__bf16)qrow[d0 + 8 + i];
            }
        }
    }

    // ---------------- Phase 1: S = scale*(Q K^T), masked, into LDS ---------
    for (int t = wave; t < L_ / 16; t += NWAVES) {
        const int k0 = t * 16;
        v8f acc = {};
        const float* krow = K + (size_t)(k0 + l16) * D_;    // B col n = K row n
        #pragma unroll
        for (int c = 0; c < 4; ++c) {
            v16bf bfrag;                                    // B 32x16: lane n = l16
            const int d0 = c * 32 + 16 * half;
            #pragma unroll
            for (int i = 0; i < 16; ++i)
                bfrag[i] = (__bf16)krow[d0 + i];
            acc = __builtin_amdgcn_wmma_f32_16x16x32_bf16(false, a_q[c], false, bfrag,
                                                          (short)0, acc, false, false);
        }
        // C/D layout: vgpr r -> row M = r + 8*half, col N = l16
        #pragma unroll
        for (int r = 0; r < 8; ++r) {
            const int row = r + 8 * half;
            const int col = k0 + l16;
            const int mm  = M[(size_t)row * L_ + col];
            s_scores[row * SSTRIDE + col] = (mm == 0) ? 1e-12f : acc[r] * scale;
        }
    }
    __syncthreads();

    // ---------------- Phase 2: softmax over each 2048-row in LDS -----------
    const int row = tid >> 4;           // 0..15
    const int j   = tid & 15;           // 16 threads per row, float4 granularity
    {
        const f4* srow = (const f4*)(s_scores + row * SSTRIDE);
        float m = -3.4e38f;
        for (int c4 = j; c4 < L_ / 4; c4 += 16) {
            const f4 x = srow[c4];
            m = fmaxf(m, fmaxf(fmaxf(x[0], x[1]), fmaxf(x[2], x[3])));
        }
        s_red[row * 16 + j] = m;
    }
    __syncthreads();
    if (j == 0) {
        float m = s_red[row * 16];
        #pragma unroll
        for (int i = 1; i < 16; ++i) m = fmaxf(m, s_red[row * 16 + i]);
        s_stat[row] = m;
    }
    __syncthreads();
    {
        const float m = s_stat[row];
        f4* srow = (f4*)(s_scores + row * SSTRIDE);
        float sum = 0.f;
        for (int c4 = j; c4 < L_ / 4; c4 += 16) {
            f4 x = srow[c4];
            f4 e;
            e[0] = __expf(x[0] - m); e[1] = __expf(x[1] - m);
            e[2] = __expf(x[2] - m); e[3] = __expf(x[3] - m);
            srow[c4] = e;                       // keep unnormalized exp for PV
            sum += (e[0] + e[1]) + (e[2] + e[3]);
        }
        s_red[row * 16 + j] = sum;
    }
    __syncthreads();
    if (j == 0) {
        float sum = 0.f;
        #pragma unroll
        for (int i = 0; i < 16; ++i) sum += s_red[row * 16 + i];
        s_stat[16 + row] = 1.0f / sum;
    }
    __syncthreads();

    // ---------------- Phase 3: stream normalized probs (non-temporal) ------
    {
        const float inv = s_stat[16 + row];
        const f4* srow = (const f4*)(s_scores + row * SSTRIDE);
        f4* prow = (f4*)(probsg + (head * L_ + (size_t)(q0 + row)) * L_);
        for (int c4 = j; c4 < L_ / 4; c4 += 16) {
            f4 p = srow[c4];
            p[0] *= inv; p[1] *= inv; p[2] *= inv; p[3] *= inv;
            __builtin_nontemporal_store(p, prow + c4);
        }
    }

    // ---------------- Phase 4: O = softmax(S) V  (normalize at the end) ----
    {
        const int d0 = wave * 16;       // each wave owns a 16-wide D column tile
        v8f acc = {};
        for (int kt = 0; kt < L_ / 32; ++kt) {
            const int kb = kt * 32;
            // A: P tile 16x32 from LDS via 4x ds_load_b128
            v16bf afrag;
            {
                const f4* srow = (const f4*)(s_scores + l16 * SSTRIDE);
                const int qi = (kb + 16 * half) >> 2;
                const f4 p0 = srow[qi];
                const f4 p1 = srow[qi + 1];
                const f4 p2 = srow[qi + 2];
                const f4 p3 = srow[qi + 3];
                #pragma unroll
                for (int i = 0; i < 4; ++i) {
                    afrag[i]      = (__bf16)p0[i];
                    afrag[i + 4]  = (__bf16)p1[i];
                    afrag[i + 8]  = (__bf16)p2[i];
                    afrag[i + 12] = (__bf16)p3[i];
                }
            }
            // B: V tile 32x16 (col n = d0+l16; half-wave selects K half)
            v16bf bfrag;
            {
                const float* vcol = V + (size_t)(kb + 16 * half) * D_ + d0 + l16;
                #pragma unroll
                for (int i = 0; i < 16; ++i)
                    bfrag[i] = (__bf16)vcol[(size_t)i * D_];
            }
            acc = __builtin_amdgcn_wmma_f32_16x16x32_bf16(false, afrag, false, bfrag,
                                                          (short)0, acc, false, false);
        }
        #pragma unroll
        for (int r = 0; r < 8; ++r) {
            const int m = r + 8 * half;
            const float o = acc[r] * s_stat[16 + m];
            __builtin_nontemporal_store(
                o, outg + (head * L_ + (size_t)(q0 + m)) * D_ + d0 + l16);
        }
    }
}

extern "C" void kernel_launch(void* const* d_in, const int* in_sizes, int n_in,
                              void* d_out, int out_size, void* d_ws, size_t ws_size,
                              hipStream_t stream) {
    const float* q    = (const float*)d_in[0];
    const float* k    = (const float*)d_in[1];
    const float* v    = (const float*)d_in[2];
    const int*   mask = (const int*)d_in[3];

    float* out   = (float*)d_out;                              // [B,H,L,D]
    float* probs = out + (size_t)B_ * H_ * L_ * D_;            // [B,H,L,L]

    const size_t shmem = (size_t)(QTILE * SSTRIDE + QTILE * 16 + 32) * sizeof(float);
    (void)hipFuncSetAttribute((const void*)sdpa_bf16_wmma_kernel,
                              hipFuncAttributeMaxDynamicSharedMemorySize, (int)shmem);

    const int nblocks = B_ * H_ * (L_ / QTILE);                // 8192
    sdpa_bf16_wmma_kernel<<<nblocks, THREADS, shmem, stream>>>(q, k, v, mask, out, probs);
}
